// MPSAE_35622458753219
// MI455X (gfx1250) — matrix operand order repaired and checked by reading, not compiled, not run on hardware
//
#include <hip/hip_runtime.h>
#include <hip/hip_bf16.h>

#define D_IN   2048
#define WIDTH  16384
#define BATCH  8192
#define KSTEPS 16
#define NSLICE (D_IN / 32)   // 64 K-slices of 32

typedef __attribute__((ext_vector_type(16))) __bf16 v16bf;
typedef __attribute__((ext_vector_type(8)))  __bf16 v8bf;
typedef __attribute__((ext_vector_type(8)))  float  v8f;

// pointer types for the async-to-LDS builtin (param types per hipcc diagnostic:
// int4 in addrspace(1) / addrspace(3), non-const)
typedef int v4i __attribute__((__vector_size__(16)));
typedef __attribute__((address_space(1))) v4i as1_v4i;
typedef __attribute__((address_space(3))) v4i as3_v4i;

// ---- workspace layout (bytes) ----
#define OFF_WN16   0ull                      // 16384*2048*2  = 64 MB  (bf16 normalized dict)
#define OFF_R      67108864ull               // 8192*2048*4   = 64 MB  (f32 residual)
#define OFF_R16    134217728ull              // 8192*2048*2   = 32 MB  (bf16 residual mirror)
#define OFF_INVN   167772160ull              // 16384*4                 (1/norm per row)
#define OFF_SLOTS  167837696ull              // 8192*8                  (argmax u64 slots)
#define OFF_CHOSEN 167903232ull              // 16*8192*4
#define OFF_COEFF  168427520ull              // 16*8192*4
// total ~161 MB

#if defined(__has_builtin)
#if __has_builtin(__builtin_amdgcn_global_load_async_to_lds_b128)
#define ASYNC_LDS 1
#endif
#endif
#ifndef ASYNC_LDS
#define ASYNC_LDS 0
#endif

__device__ __forceinline__ v16bf combine16(v8bf lo, v8bf hi) {
    v16bf r;
#pragma unroll
    for (int i = 0; i < 8; ++i) { r[i] = lo[i]; r[i + 8] = hi[i]; }
    return r;
}

__device__ __forceinline__ unsigned long long shflxor_u64(unsigned long long v, int m) {
    unsigned int lo = (unsigned int)v, hi = (unsigned int)(v >> 32);
    lo = __shfl_xor(lo, m, 32);
    hi = __shfl_xor(hi, m, 32);
    return ((unsigned long long)hi << 32) | lo;
}

__device__ __forceinline__ void async_copy32(const __bf16* g, __bf16* l) {
#if ASYNC_LDS
    as1_v4i* gp = (as1_v4i*)(uintptr_t)g;
    as3_v4i* lp = (as3_v4i*)(unsigned int)(uintptr_t)l;
    __builtin_amdgcn_global_load_async_to_lds_b128(gp, lp, 0, 0);
    __builtin_amdgcn_global_load_async_to_lds_b128(gp + 1, lp + 1, 0, 0);
#else
    const uint4* src = (const uint4*)g;
    uint4* dst = (uint4*)l;
    dst[0] = src[0];
    dst[1] = src[1];
#endif
}

__device__ __forceinline__ void wait_async0() {
#if ASYNC_LDS
#if __has_builtin(__builtin_amdgcn_s_wait_asynccnt)
    __builtin_amdgcn_s_wait_asynccnt(0);
#else
    asm volatile("s_wait_asynccnt 0" ::: "memory");
#endif
#endif
}

// stage one 32-wide K slice of A (256 rows of R16) and B (256 cols of Wn) into LDS.
// 512 threads: 2 threads per row/col, 32 B each (2x b128).
__device__ __forceinline__ void stage_slice(const __bf16* __restrict__ R16,
                                            const __bf16* __restrict__ Wn16,
                                            __bf16* bufA, __bf16* bufB,
                                            int row0, int n0, int kb, int tid) {
    const int r = tid >> 1, half = tid & 1;
    async_copy32(R16 + (size_t)(row0 + r) * D_IN + kb + half * 16, bufA + r * 48 + half * 16);
    async_copy32(Wn16 + (size_t)(n0 + r) * D_IN + kb + half * 16, bufB + r * 48 + half * 16);
}

// ---- prep: row norms of W, bf16 normalized dictionary ----
__global__ __launch_bounds__(256)
void prep_norm_kernel(const float* __restrict__ W, __bf16* __restrict__ Wn16,
                      float* __restrict__ invn) {
    const int j = blockIdx.x;
    const float* w = W + (size_t)j * D_IN;
    float ss = 0.f;
    for (int d = threadIdx.x; d < D_IN; d += 256) { float v = w[d]; ss += v * v; }
#pragma unroll
    for (int m = 16; m >= 1; m >>= 1) ss += __shfl_xor(ss, m, 32);
    __shared__ float red[8];
    __shared__ float s_inv;
    const int wid = threadIdx.x >> 5, lane = threadIdx.x & 31;
    if (lane == 0) red[wid] = ss;
    __syncthreads();
    if (threadIdx.x == 0) {
        float t = red[0] + red[1] + red[2] + red[3] + red[4] + red[5] + red[6] + red[7];
        float inv = 1.0f / fmaxf(sqrtf(t), 1e-12f);
        invn[j] = inv;
        s_inv = inv;
    }
    __syncthreads();
    const float inv = s_inv;
    __bf16* o = Wn16 + (size_t)j * D_IN;
    for (int d = threadIdx.x; d < D_IN; d += 256) o[d] = (__bf16)(w[d] * inv);
}

// ---- init residual: R = x (f32), R16 = bf16(x) ----
__global__ __launch_bounds__(256)
void init_R_kernel(const float* __restrict__ x, float* __restrict__ R,
                   __bf16* __restrict__ R16) {
    const size_t n = (size_t)BATCH * D_IN;
    for (size_t i = (size_t)blockIdx.x * blockDim.x + threadIdx.x; i < n;
         i += (size_t)gridDim.x * blockDim.x) {
        float v = x[i];
        R[i] = v;
        R16[i] = (__bf16)v;
    }
}

// ---- zero fill (float4 granularity, grid-stride) ----
__global__ __launch_bounds__(256)
void fill_zero4(float4* __restrict__ p, size_t n4) {
    for (size_t i = (size_t)blockIdx.x * blockDim.x + threadIdx.x; i < n4;
         i += (size_t)gridDim.x * blockDim.x)
        p[i] = make_float4(0.f, 0.f, 0.f, 0.f);
}

// ---- fused score GEMM + relu + per-row argmax ----
// block: 512 threads = 16 waves (4x4), tile 256 rows x 256 cols (square = min L2 traffic)
// wave tile: 64x64 via 4x4 WMMA 16x16x32 bf16; double-buffered async LDS staging
__global__ __launch_bounds__(512)
void score_argmax_kernel(const __bf16* __restrict__ R16, const __bf16* __restrict__ Wn16,
                         unsigned long long* __restrict__ slots) {
    __shared__ __bf16 bufA[2][256 * 48];       // 256 rows x 32 k (stride 48), x2 buffers
    __shared__ __bf16 bufB[2][256 * 48];       // 256 cols x 32 k (row-major = Wn layout)
    __shared__ unsigned long long lslot[256];  // per-local-row packed (score, ~col)

    const int tid  = threadIdx.x;
    const int lane = tid & 31, wid = tid >> 5;
    const int rg = wid >> 2, cg = wid & 3;     // 4 row-groups x 4 col-groups of 64
    const int row0 = blockIdx.y * 256;
    const int n0   = blockIdx.x * 256;

    if (tid < 256) lslot[tid] = 0ull;

    v8f acc[4][4];
#pragma unroll
    for (int i = 0; i < 4; ++i)
#pragma unroll
        for (int j = 0; j < 4; ++j)
#pragma unroll
            for (int e = 0; e < 8; ++e) acc[i][j][e] = 0.f;

    // prologue: stage slice 0
    stage_slice(R16, Wn16, bufA[0], bufB[0], row0, n0, 0, tid);

    const int koff = (lane >> 4) * 8;    // unused (table-order K base); both fragments use klow
    const int klow = (lane >> 4) * 16;   // A/B-fragment K base, contiguous k = 16h + s

    for (int it = 0; it < NSLICE; ++it) {
        wait_async0();        // my slice-`it` async writes have landed in LDS
        __syncthreads();      // everyone's slice `it` visible; buffer (it+1)&1 free
        if (it + 1 < NSLICE)
            stage_slice(R16, Wn16, bufA[(it + 1) & 1], bufB[(it + 1) & 1],
                        row0, n0, (it + 1) * 32, tid);

        const __bf16* lA = bufA[it & 1];
        const __bf16* lB = bufB[it & 1];
        v16bf afrag[4], bfrag[4];
#pragma unroll
        for (int mi = 0; mi < 4; ++mi) {
            const int r = rg * 64 + mi * 16 + (lane & 15);
            const v8bf* p = (const v8bf*)(lA + r * 48 + klow);
            afrag[mi] = combine16(p[0], p[1]);  // contiguous K klow..klow+15
        }
#pragma unroll
        for (int ni = 0; ni < 4; ++ni) {
            const int c = cg * 64 + ni * 16 + (lane & 15);
            const v8bf* p = (const v8bf*)(lB + c * 48 + klow);
            bfrag[ni] = combine16(p[0], p[1]);  // contiguous K klow..klow+15
        }
#pragma unroll
        for (int mi = 0; mi < 4; ++mi)
#pragma unroll
            for (int ni = 0; ni < 4; ++ni)
                acc[mi][ni] = __builtin_amdgcn_wmma_f32_16x16x32_bf16(
                    false, afrag[mi], false, bfrag[ni], (short)0, acc[mi][ni], false, false);
    }

    // epilogue: relu + per-row argmax. C/D layout: VGPR r -> row r (lanes 0-15) / r+8 (16-31)
#pragma unroll
    for (int mi = 0; mi < 4; ++mi) {
#pragma unroll
        for (int r = 0; r < 8; ++r) {
            unsigned long long best = 0ull;
#pragma unroll
            for (int ni = 0; ni < 4; ++ni) {
                float v = acc[mi][ni][r];
                v = v > 0.f ? v : 0.f;
                const unsigned int gcol =
                    (unsigned int)(n0 + cg * 64 + ni * 16 + (lane & 15));
                const unsigned long long p =
                    ((unsigned long long)__float_as_uint(v) << 32) | (0xFFFFFFFFu - gcol);
                best = p > best ? p : best;
            }
#pragma unroll
            for (int m = 1; m < 16; m <<= 1) {
                unsigned long long o = shflxor_u64(best, m);
                best = o > best ? o : best;
            }
            if ((lane & 15) == 0) {
                const int lr = rg * 64 + mi * 16 + r + ((lane >> 4) << 3);
                atomicMax(&lslot[lr], best);
            }
        }
    }
    __syncthreads();
    if (tid < 256) atomicMax(&slots[row0 + tid], lslot[tid]);
}

// ---- per-row: exact f32 coefficient of the winner, residual update, bookkeeping ----
__global__ __launch_bounds__(256)
void update_kernel(const float* __restrict__ W, const float* __restrict__ invn,
                   float* __restrict__ R, __bf16* __restrict__ R16,
                   unsigned long long* __restrict__ slots, float* __restrict__ code,
                   int* __restrict__ chosen, float* __restrict__ coeffs, int step) {
    const int row = blockIdx.x;
    const unsigned long long p = slots[row];
    unsigned int col = 0xFFFFFFFFu - (unsigned int)(p & 0xFFFFFFFFull);
    if (col >= WIDTH) col = 0;
    const float iv = invn[col];
    const float* w = W + (size_t)col * D_IN;
    float* r = R + (size_t)row * D_IN;

    float part = 0.f;
    for (int d = threadIdx.x; d < D_IN; d += 256) part += r[d] * w[d];
#pragma unroll
    for (int m = 16; m >= 1; m >>= 1) part += __shfl_xor(part, m, 32);
    __shared__ float red[8];
    __shared__ float s_c;
    const int wid = threadIdx.x >> 5, lane = threadIdx.x & 31;
    if (lane == 0) red[wid] = part;
    __syncthreads();
    if (threadIdx.x == 0) {
        float t = red[0] + red[1] + red[2] + red[3] + red[4] + red[5] + red[6] + red[7];
        float c = t * iv;               // exact f32 score of the chosen atom
        c = c > 1e-8f ? c : 0.f;        // relu + active threshold (matches reference)
        s_c = c;
        slots[row] = 0ull;              // reset for next step
        chosen[step * BATCH + row] = (int)col;
        coeffs[step * BATCH + row] = c;
        code[(size_t)row * WIDTH + col] += c;
    }
    __syncthreads();
    const float c = s_c;
    __bf16* r16 = R16 + (size_t)row * D_IN;
    for (int d = threadIdx.x; d < D_IN; d += 256) {
        const float nv = r[d] - c * (w[d] * iv);
        r[d] = nv;
        r16[d] = (__bf16)nv;
    }
}

// ---- sparse decode: x_hat = bias + sum_k coeff_k * W[chosen_k] ----
__global__ __launch_bounds__(256)
void decode_kernel(const float* __restrict__ W, const float* __restrict__ bias,
                   const int* __restrict__ chosen, const float* __restrict__ coeffs,
                   float* __restrict__ xhat) {
    const int row = blockIdx.x;
    float acc[8];
#pragma unroll
    for (int i = 0; i < 8; ++i) acc[i] = bias[threadIdx.x + 256 * i];
    for (int k = 0; k < KSTEPS; ++k) {
        const float c = coeffs[k * BATCH + row];
        const int j = chosen[k * BATCH + row];
        const float* w = W + (size_t)j * D_IN;
#pragma unroll
        for (int i = 0; i < 8; ++i) acc[i] += c * w[threadIdx.x + 256 * i];
    }
    float* o = xhat + (size_t)row * D_IN;
#pragma unroll
    for (int i = 0; i < 8; ++i) o[threadIdx.x + 256 * i] = acc[i];
}

extern "C" void kernel_launch(void* const* d_in, const int* in_sizes, int n_in,
                              void* d_out, int out_size, void* d_ws, size_t ws_size,
                              hipStream_t stream) {
    const float* x    = (const float*)d_in[0];  // [8192, 2048]
    const float* W    = (const float*)d_in[1];  // [16384, 2048]
    const float* bias = (const float*)d_in[2];  // [2048]

    char* ws = (char*)d_ws;
    __bf16* Wn16 = (__bf16*)(ws + OFF_WN16);
    float* R     = (float*)(ws + OFF_R);
    __bf16* R16  = (__bf16*)(ws + OFF_R16);
    float* invn  = (float*)(ws + OFF_INVN);
    unsigned long long* slots = (unsigned long long*)(ws + OFF_SLOTS);
    int* chosen  = (int*)(ws + OFF_CHOSEN);
    float* coeffs = (float*)(ws + OFF_COEFF);

    float* code = (float*)d_out;                        // [8192, 16384]
    float* xhat = code + (size_t)BATCH * WIDTH;         // [8192, 2048]

    prep_norm_kernel<<<WIDTH, 256, 0, stream>>>(W, Wn16, invn);
    init_R_kernel<<<4096, 256, 0, stream>>>(x, R, R16);
    fill_zero4<<<16384, 256, 0, stream>>>((float4*)code, (size_t)BATCH * WIDTH / 4);
    fill_zero4<<<64, 256, 0, stream>>>((float4*)slots, (size_t)4096);

    dim3 sg(WIDTH / 256, BATCH / 256);   // 64 x 32 blocks, 512 threads each
    for (int k = 0; k < KSTEPS; ++k) {
        score_argmax_kernel<<<sg, 512, 0, stream>>>(R16, Wn16, slots);
        update_kernel<<<BATCH, 256, 0, stream>>>(W, invn, R, R16, slots, code, chosen,
                                                 coeffs, k);
    }
    decode_kernel<<<BATCH, 256, 0, stream>>>(W, bias, chosen, coeffs, xhat);
}
